// MultiHeadSelfAttentionTENER_77077483094286
// MI455X (gfx1250) — matrix operand order repaired and checked by reading, not compiled, not run on hardware
//
#include <hip/hip_runtime.h>
#include <hip/hip_bf16.h>
#include <math.h>

typedef __attribute__((ext_vector_type(16))) __bf16 v16bf;
typedef __attribute__((ext_vector_type(8)))  float  v8f;
typedef __attribute__((ext_vector_type(4)))  unsigned int v4u;
typedef __attribute__((ext_vector_type(8)))  int v8i;
typedef __attribute__((ext_vector_type(4)))  int v4i;

#define Bb   8
#define Tt   1024
#define Dd   512
#define Hh   8
#define DHd  64
#define PE_ROWS (2*Tt + 16)   // pad past 2T for band edge (l can touch 2T)

#if defined(__gfx1250__) && __has_builtin(__builtin_amdgcn_tensor_load_to_lds)
#define HAS_TDM 1
#else
#define HAS_TDM 0
#endif

// A-fragment K index for 16x32 bf16 A (contiguous order, matches B fills):
// lanes 0-15: row M=lane, K in {0-15}; lanes 16-31: row M=lane-16, K in {16-31}
__device__ __forceinline__ int ka_map(int e, int hi) {
  return e + (hi << 4);
}

// ---------------- PE table: pe[l][d], l in [0,2T), bf16, zero-padded ----------------
__global__ void pe_kernel(__bf16* pebf) {
  int idx = blockIdx.x * blockDim.x + threadIdx.x;
  if (idx >= PE_ROWS * DHd) return;
  int l = idx >> 6;
  int d = idx & 63;
  float v = 0.f;
  if (l < 2 * Tt) {
    float pos = (float)(l - Tt);
    int j = d & 31;
    float div = __expf((float)j * (-9.210340371976184f / 31.0f));  // ln(1e4)/(half-1)
    float ang = pos * div;
    v = (d < 32) ? __sinf(ang) : __cosf(ang);
  }
  pebf[idx] = (__bf16)v;
}

// ---------------- f32 -> bf16 elementwise convert (for key_in) ----------------
__global__ void cvt_kernel(const float* __restrict__ in, __bf16* __restrict__ out, int n) {
  int i = blockIdx.x * blockDim.x + threadIdx.x;
  if (i < n) out[i] = (__bf16)in[i];
}

// ---------------- GEMM: f32 A [MxK] @ f32 W [KxN] + bias -> bf16 out ----------------
__global__ void gemm_f32A_bf16out(const float* __restrict__ A, const float* __restrict__ W,
                                  const float* __restrict__ bias, const float* __restrict__ addvec,
                                  __bf16* __restrict__ out0, __bf16* out1,
                                  int M, int N, int K) {
  int lane = threadIdx.x;
  int hi = lane >> 4, lo = lane & 15;
  int col0 = (blockIdx.x * blockDim.y + threadIdx.y) * 16;
  int row0 = blockIdx.y * 16;
  v8f c = {};
  for (int kb = 0; kb < K; kb += 32) {
    v16bf a, b;
#pragma unroll
    for (int e = 0; e < 16; ++e) {
      int ka = ka_map(e, hi);
      a[e] = (__bf16)A[(size_t)(row0 + lo) * K + kb + ka];
      int kw = e + (hi << 4);
      b[e] = (__bf16)W[(size_t)(kb + kw) * N + col0 + lo];
    }
    c = __builtin_amdgcn_wmma_f32_16x16x32_bf16(false, a, false, b, (short)0, c, false, false);
  }
#pragma unroll
  for (int r = 0; r < 8; ++r) {
    int m = row0 + r + (hi << 3);
    int n = col0 + lo;
    float v = c[r] + bias[n];
    out0[(size_t)m * N + n] = (__bf16)v;
    if (out1) out1[(size_t)m * N + n] = (__bf16)(v + addvec[n]);
  }
}

// ---------------- GEMM: bf16 A [MxK] @ f32 W [KxN] + bias -> f32 out ----------------
__global__ void gemm_bf16A_f32out(const __bf16* __restrict__ A, const float* __restrict__ W,
                                  const float* __restrict__ bias, float* __restrict__ out,
                                  int M, int N, int K) {
  int lane = threadIdx.x;
  int hi = lane >> 4, lo = lane & 15;
  int col0 = (blockIdx.x * blockDim.y + threadIdx.y) * 16;
  int row0 = blockIdx.y * 16;
  v8f c = {};
  for (int kb = 0; kb < K; kb += 32) {
    v16bf a, b;
#pragma unroll
    for (int e = 0; e < 16; ++e) {
      int ka = ka_map(e, hi);
      a[e] = A[(size_t)(row0 + lo) * K + kb + ka];
      int kw = e + (hi << 4);
      b[e] = (__bf16)W[(size_t)(kb + kw) * N + col0 + lo];
    }
    c = __builtin_amdgcn_wmma_f32_16x16x32_bf16(false, a, false, b, (short)0, c, false, false);
  }
#pragma unroll
  for (int r = 0; r < 8; ++r) {
    int m = row0 + r + (hi << 3);
    int n = col0 + lo;
    out[(size_t)m * N + n] = c[r] + bias[n];
  }
}

// ---------------- TDM: DMA a 32x64 bf16 tile (row stride Dd elems) into LDS ----------
#if HAS_TDM
__device__ __forceinline__ void tdm_load_tile(const __bf16* gsrc, void* ldst) {
  unsigned long long ga = (unsigned long long)gsrc;
  v4u g0;
  g0[0] = 1u;                                   // count=1, user mode, no gather
  g0[1] = (unsigned)(unsigned long long)ldst;   // lds byte address
  g0[2] = (unsigned)ga;                         // global_addr[31:0]
  g0[3] = ((unsigned)(ga >> 32) & 0x01FFFFFFu) | (2u << 30);  // addr[56:32] | type=2
  v8i g1;
  g1[0] = (1 << 16);                 // workgroup_mask=0, data_size=1 (2 bytes)
  g1[1] = (DHd & 0xFFFF) << 16;      // tensor_dim0 = 64 (lo16 at bits 63:48)
  g1[2] = (Tt & 0xFFFF) << 16;       // tensor_dim0 hi=0; tensor_dim1 = T (lo16)
  g1[3] = (DHd & 0xFFFF) << 16;      // tensor_dim1 hi=0; tile_dim0 = 64
  g1[4] = 32;                        // tile_dim1 = 32, tile_dim2 = 0
  g1[5] = Dd;                        // tensor_dim0_stride = 512 (lo32)
  g1[6] = 0;                         // stride hi16 = 0; tensor_dim1_stride lo16 = 0
  g1[7] = 0;
  v4i z4 = {0, 0, 0, 0};
  v8i z8 = {0, 0, 0, 0, 0, 0, 0, 0};
  // clang-23 / therock-10.0 arity: (g0, g1, g2, g3, g4, cpol)
  __builtin_amdgcn_tensor_load_to_lds(g0, g1, z4, z4, z8, 0);
}
#endif

__device__ __forceinline__ void stage_tile_coop(const __bf16* g, __bf16 (*lds)[DHd], int tid) {
  // fallback: cooperative dword copy of 32x64 bf16 tile (1024 dwords, 128 threads)
  for (int w = tid; w < 32 * 32; w += 128) {
    int row = w >> 5, cw = w & 31;
    *(unsigned*)(&lds[row][cw * 2]) = *(const unsigned*)(g + (size_t)row * Dd + cw * 2);
  }
}

// ---------------- Fused TENER flash attention ----------------
// rel[i,j] = (q_i + v_bias_h) . pe[T+j-i]  (skew trick folded into Q' = Q + v_bias)
// 4 waves / block share (b,h); K/V tiles DMA'd by TDM into double-buffered LDS.
__global__ void attn_kernel(const __bf16* __restrict__ qbf,  // [B,T,D] = q
                            const __bf16* __restrict__ qpb,  // [B,T,D] = q + v_bias
                            const __bf16* __restrict__ kbf,  // [B,T,D]
                            const __bf16* __restrict__ vbf,  // [B,T,D]
                            const __bf16* __restrict__ pebf, // [PE_ROWS,64]
                            const int*    __restrict__ mask, // [B,T]
                            __bf16* __restrict__ xbf) {      // [B,T,D] attn@v
  __shared__ float  pbuf[4][16][33];      // per-wave P staging (16x32, padded)
  __shared__ __bf16 ldsK[2][32][DHd];     // double-buffered K tile (32 keys x 64 dims)
  __shared__ __bf16 ldsV[2][32][DHd];     // double-buffered V tile
  int lane = threadIdx.x, wy = threadIdx.y;
  int hi = lane >> 4, lo = lane & 15;
  int tid = wy * 32 + lane;
  int qb = blockIdx.x * 4 + wy;
  int bh = blockIdx.y;
  int b = bh >> 3, h = bh & 7;
  int i0 = qb * 16;
  size_t rowQ = ((size_t)b * Tt + i0 + lo) * Dd + h * 64;
  size_t tileBase0 = ((size_t)b * Tt) * Dd + h * 64;

  // preload Q fragments (content) and Q' fragments (relative), K-halves 0-31 / 32-63
  v16bf aQ0, aQ1, aP0, aP1;
#pragma unroll
  for (int e = 0; e < 16; ++e) {
    int ka = ka_map(e, hi);
    aQ0[e] = qbf[rowQ + ka];
    aQ1[e] = qbf[rowQ + 32 + ka];
    aP0[e] = qpb[rowQ + ka];
    aP1[e] = qpb[rowQ + 32 + ka];
  }

  float rmax[8], rsum[8];
  v8f acc[4];
  v8f vzero = {};
#pragma unroll
  for (int r = 0; r < 8; ++r) { rmax[r] = -INFINITY; rsum[r] = 0.f; }
#pragma unroll
  for (int f = 0; f < 4; ++f) acc[f] = vzero;

  // ---- prologue: stage tile for j0=0 into buffer 0
#if HAS_TDM
  if (wy == 0) {
    tdm_load_tile(kbf + tileBase0, &ldsK[0][0][0]);
    tdm_load_tile(vbf + tileBase0, &ldsV[0][0][0]);
    __builtin_amdgcn_s_wait_tensorcnt(0);
  }
#else
  stage_tile_coop(kbf + tileBase0, ldsK[0], tid);
  stage_tile_coop(vbf + tileBase0, ldsV[0], tid);
#endif
  __syncthreads();

  int cur = 0;
  for (int j0 = 0; j0 < Tt; j0 += 32) {
    // ---- async prefetch of next key tile into the other buffer
    if (j0 + 32 < Tt) {
      size_t nb = tileBase0 + (size_t)(j0 + 32) * Dd;
#if HAS_TDM
      if (wy == 0) {
        tdm_load_tile(kbf + nb, &ldsK[cur ^ 1][0][0]);
        tdm_load_tile(vbf + nb, &ldsV[cur ^ 1][0][0]);
      }
#else
      stage_tile_coop(kbf + nb, ldsK[cur ^ 1], tid);
      stage_tile_coop(vbf + nb, ldsV[cur ^ 1], tid);
#endif
    }

    // ---- content logits: two 16x16 tiles (keys j0..j0+15 and j0+16..j0+31)
    v8f s[2];
#pragma unroll
    for (int t2 = 0; t2 < 2; ++t2) {
      v16bf bk0, bk1;
#pragma unroll
      for (int e = 0; e < 16; ++e) {
        int kw = e + (hi << 4);
        bk0[e] = ldsK[cur][t2 * 16 + lo][kw];       // B = K^T : (kdim, key)
        bk1[e] = ldsK[cur][t2 * 16 + lo][32 + kw];
      }
      v8f cc = vzero;
      cc = __builtin_amdgcn_wmma_f32_16x16x32_bf16(false, aQ0, false, bk0, (short)0, cc, false, false);
      cc = __builtin_amdgcn_wmma_f32_16x16x32_bf16(false, aQ1, false, bk1, (short)0, cc, false, false);
      s[t2] = cc;
    }

    // ---- relative band: qr strip [16 x 48], columns l = l0 + c
    int l0 = Tt + j0 - i0 - 15;           // in [1, 2T-47]
    v8f qrt[3];
#pragma unroll
    for (int t = 0; t < 3; ++t) {
      int l = l0 + t * 16 + lo;
      v16bf bp0, bp1;
#pragma unroll
      for (int e = 0; e < 16; ++e) {
        int kw = e + (hi << 4);
        bp0[e] = pebf[(size_t)l * 64 + kw];       // B = PE^T
        bp1[e] = pebf[(size_t)l * 64 + 32 + kw];
      }
      v8f cc = vzero;
      cc = __builtin_amdgcn_wmma_f32_16x16x32_bf16(false, aP0, false, bp0, (short)0, cc, false, false);
      cc = __builtin_amdgcn_wmma_f32_16x16x32_bf16(false, aP1, false, bp1, (short)0, cc, false, false);
      qrt[t] = cc;
    }

    // ---- skew gather: rel(m,n) = qr[m][15 + n - m (+16)] ; add + mask
    int msk0 = mask[b * Tt + j0 + lo];
    int msk1 = mask[b * Tt + j0 + 16 + lo];
#pragma unroll
    for (int r = 0; r < 8; ++r) {
      int m = r + (hi << 3);
      int c0 = 15 + lo - m;               // 0..30  -> tiles 0/1
      int c1 = 31 + lo - m;               // 16..46 -> tiles 1/2
      int sl0 = (lane & 16) | (c0 & 15);
      int sl1 = (lane & 16) | (c1 & 15);
      float g00 = __shfl(qrt[0][r], sl0, 32);
      float g01 = __shfl(qrt[1][r], sl0, 32);
      float g10 = __shfl(qrt[1][r], sl1, 32);
      float g11 = __shfl(qrt[2][r], sl1, 32);
      float rel0 = (c0 < 16) ? g00 : g01;
      float rel1 = (c1 < 32) ? g10 : g11;
      float v0 = s[0][r] + rel0;
      float v1 = s[1][r] + rel1;
      if (!msk0) v0 = -1e30f;
      if (!msk1) v1 = -1e30f;
      s[0][r] = v0;
      s[1][r] = v1;
    }

    // ---- online softmax (row = r + 8*hi, cols across 16 lanes of the half)
    float nscale[8];
#pragma unroll
    for (int r = 0; r < 8; ++r) {
      float mx = fmaxf(s[0][r], s[1][r]);
#pragma unroll
      for (int off = 1; off < 16; off <<= 1) mx = fmaxf(mx, __shfl_xor(mx, off, 32));
      float nm = fmaxf(rmax[r], mx);
      float sc = __expf(rmax[r] - nm);
      rmax[r] = nm;
      float p0 = __expf(s[0][r] - nm);
      float p1 = __expf(s[1][r] - nm);
      s[0][r] = p0;
      s[1][r] = p1;
      float ps = p0 + p1;
#pragma unroll
      for (int off = 1; off < 16; off <<= 1) ps += __shfl_xor(ps, off, 32);
      rsum[r] = rsum[r] * sc + ps;
      nscale[r] = sc;
    }
#pragma unroll
    for (int f = 0; f < 4; ++f)
#pragma unroll
      for (int r = 0; r < 8; ++r) acc[f][r] = acc[f][r] * nscale[r];

    // ---- stage P (C-layout f32) -> LDS -> A-fragment bf16 (same wave; DS in-order)
#pragma unroll
    for (int r = 0; r < 8; ++r) {
      pbuf[wy][r + (hi << 3)][lo]      = s[0][r];
      pbuf[wy][r + (hi << 3)][lo + 16] = s[1][r];
    }
    v16bf aPfrag;
#pragma unroll
    for (int e = 0; e < 16; ++e) {
      int ka = ka_map(e, hi);
      aPfrag[e] = (__bf16)pbuf[wy][lo][ka];
    }

    // ---- AV: acc[16x64] += P[16x32] @ V[32x64]
#pragma unroll
    for (int f = 0; f < 4; ++f) {
      v16bf bV;
#pragma unroll
      for (int e = 0; e < 16; ++e) {
        int kw = e + (hi << 4);           // key row within the 32-block
        bV[e] = ldsV[cur][kw][f * 16 + lo];
      }
      acc[f] = __builtin_amdgcn_wmma_f32_16x16x32_bf16(false, aPfrag, false, bV, (short)0, acc[f], false, false);
    }

    // ---- close the double-buffer stage: DMA done + all waves done with buffers
#if HAS_TDM
    if (wy == 0) __builtin_amdgcn_s_wait_tensorcnt(0);
#endif
    __syncthreads();
    cur ^= 1;
  }

  // ---- epilogue: normalize and store x (bf16) in [B,T,H*64] layout
#pragma unroll
  for (int f = 0; f < 4; ++f)
#pragma unroll
    for (int r = 0; r < 8; ++r) {
      float o = acc[f][r] / rsum[r];
      size_t idx = ((size_t)b * Tt + i0 + r + (hi << 3)) * Dd + h * 64 + f * 16 + lo;
      xbf[idx] = (__bf16)o;
    }
}

extern "C" void kernel_launch(void* const* d_in, const int* in_sizes, int n_in,
                              void* d_out, int out_size, void* d_ws, size_t ws_size,
                              hipStream_t stream) {
  const float* query  = (const float*)d_in[0];
  const float* key_in = (const float*)d_in[1];
  const float* value  = (const float*)d_in[2];
  const int*   mask   = (const int*)  d_in[3];
  const float* Wq     = (const float*)d_in[4];
  const float* bq     = (const float*)d_in[5];
  const float* Wv     = (const float*)d_in[6];
  const float* bv     = (const float*)d_in[7];
  const float* Wo     = (const float*)d_in[8];
  const float* bo     = (const float*)d_in[9];
  const float* v_bias = (const float*)d_in[10];   // [H*DH] flat == per-column bias

  const size_t nBTD = (size_t)Bb * Tt * Dd;       // 4,194,304 elements
  char* ws = (char*)d_ws;
  __bf16* qbf  = (__bf16*)(ws + 0 * nBTD * 2);
  __bf16* qpb  = (__bf16*)(ws + 1 * nBTD * 2);
  __bf16* kbf  = (__bf16*)(ws + 2 * nBTD * 2);
  __bf16* vbf  = (__bf16*)(ws + 3 * nBTD * 2);
  __bf16* xbf  = (__bf16*)(ws + 4 * nBTD * 2);
  __bf16* pebf = (__bf16*)(ws + 5 * nBTD * 2);    // (2T+16)*64 bf16

  // 1) positional-encoding table (bf16, zero-padded band edge)
  {
    int n = PE_ROWS * DHd;
    pe_kernel<<<dim3((n + 255) / 256), dim3(256), 0, stream>>>(pebf);
  }
  // 2) key -> bf16 (reference does NOT project keys)
  cvt_kernel<<<dim3((unsigned)((nBTD + 255) / 256)), dim3(256), 0, stream>>>(key_in, kbf, (int)nBTD);
  // 3) Q projection (+ folded v_bias copy) and V projection, WMMA GEMMs
  {
    dim3 grid(Dd / 16 / 4, (Bb * Tt) / 16);   // (8, 512)
    dim3 block(32, 4);
    gemm_f32A_bf16out<<<grid, block, 0, stream>>>(query, Wq, bq, v_bias, qbf, qpb,
                                                  Bb * Tt, Dd, Dd);
    gemm_f32A_bf16out<<<grid, block, 0, stream>>>(value, Wv, bv, nullptr, vbf, nullptr,
                                                  Bb * Tt, Dd, Dd);
  }
  // 4) fused TENER flash attention (TDM double-buffered K/V tiles)
  {
    dim3 grid(Tt / 16 / 4, Bb * Hh);          // (16, 64)
    dim3 block(32, 4);
    attn_kernel<<<grid, block, 0, stream>>>(qbf, qpb, kbf, vbf, pebf, mask, xbf);
  }
  // 5) output projection -> f32
  {
    dim3 grid(Dd / 16 / 4, (Bb * Tt) / 16);
    dim3 block(32, 4);
    gemm_bf16A_f32out<<<grid, block, 0, stream>>>(xbf, Wo, bo, (float*)d_out,
                                                  Bb * Tt, Dd, Dd);
  }
}
